// ActorNetwork_47880295416004
// MI455X (gfx1250) — compile-verified, matrix operand order checked by reading, not run on hardware
//
#include <hip/hip_runtime.h>

// ---------------------------------------------------------------------------
// CDNA5 / gfx1250 implementation.
// 4 big GEMMs (~310 GFLOP) on v_wmma_f32_16x16x32_bf16, with B pre-transposed
// so both LDS tiles are plain row-strided 2-D copies -> staged by the Tensor
// Data Mover (tensor_load_to_lds, TENSORcnt) when the builtin is available,
// double-buffered in LDS. Small softmax/LN/reduction pieces are VALU kernels.
// ---------------------------------------------------------------------------

typedef __attribute__((ext_vector_type(16))) __bf16 v16bf;
typedef __attribute__((ext_vector_type(8)))  float  v8f;
typedef __attribute__((ext_vector_type(4)))  unsigned int v4u;
typedef __attribute__((ext_vector_type(4)))  int v4i;
typedef __attribute__((ext_vector_type(8)))  int v8i;

#define E_DIM 1536
#define H_NUM 12
#define D_DIM 128
#define N_B   16
#define L_SEQ 1024
#define M_ROWS (N_B * L_SEQ)      // 16384
#define N3    (3 * E_DIM)         // 4608

// ---- workspace layout (bytes, all 256B aligned) ----
#define OFF_XB     ((size_t)0)                       // 16384x1536 bf16 = 50331648
#define OFF_WCAT   (OFF_XB   + (size_t)50331648)     // 4608x1536 bf16 (B^T) = 14155776
#define OFF_WOB    (OFF_WCAT + (size_t)14155776)     // 1536x1536 bf16 (B^T) = 4718592
#define OFF_QKV    (OFF_WOB  + (size_t)4718592)      // 16384x4608 f32 = 301989888
#define OFF_ATTNB  (OFF_QKV  + (size_t)301989888)    // 16384x1536 bf16 = 50331648
#define OFF_OBUF   (OFF_ATTNB+ (size_t)50331648)     // 16384x1536 f32 = 100663296
#define OFF_OUT2   (OFF_OBUF + (size_t)100663296)    // 16384x1536 f32 = 100663296
#define OFF_EST    (OFF_OUT2 + (size_t)100663296)    // 16x1536 f32    = 98304

// ---- TDM availability / flavor detection ----
#if __has_builtin(__builtin_amdgcn_tensor_load_to_lds) && \
    __has_builtin(__builtin_amdgcn_s_wait_tensorcnt)
#define TDM_AVAIL 1
#if __has_include(<hip/amd_detail/amd_gfx1250_TDM.h>)
#define TDM_6ARG 1
#else
#define TDM_6ARG 0
#endif
#else
#define TDM_AVAIL 0
#endif

__device__ __forceinline__ unsigned short f2bf(float f) {
  union { float f; unsigned int u; } v; v.f = f;
  unsigned int u = v.u;
  unsigned int r = u + 0x7FFFu + ((u >> 16) & 1u);   // round to nearest even
  return (unsigned short)(r >> 16);
}

#if TDM_AVAIL
// Issue one TDM 2-D tile load: `rows` rows of `rowBytes` bytes, global row
// stride `strideBytes`, into LDS at byte offset `ldsOff` (compacted rows).
// D# packing per CDNA5 ISA ch.8 (data_size = 1 byte).
__device__ __forceinline__ void tdm_load_tile(unsigned ldsOff, const void* g,
                                              unsigned rowBytes, unsigned rows,
                                              unsigned strideBytes) {
  unsigned long long ga = (unsigned long long)g;
  v4u g0;
  g0.x = 1u;                                           // count=1, user desc
  g0.y = ldsOff;                                       // lds_addr
  g0.z = (unsigned)ga;                                 // global_addr[31:0]
  g0.w = ((unsigned)(ga >> 32) & 0x01FFFFFFu)          // global_addr[56:32]
         | 0x80000000u;                                // type=2 ("image")
  v8i g1;
  g1[0] = 0;                                           // wg_mask=0, data_size=1B
  g1[1] = (int)((strideBytes & 0xFFFFu) << 16);        // tensor_dim0 lo16
  g1[2] = (int)(((strideBytes >> 16) & 0xFFFFu)        // tensor_dim0 hi16
                | ((rows & 0xFFFFu) << 16));           // tensor_dim1 lo16
  g1[3] = (int)(((rows >> 16) & 0xFFFFu)               // tensor_dim1 hi16
                | ((rowBytes & 0xFFFFu) << 16));       // tile_dim0
  g1[4] = (int)(rows & 0xFFFFu);                       // tile_dim1 (tile_dim2=0)
  g1[5] = (int)strideBytes;                            // tensor_dim0_stride lo32
  g1[6] = 0;                                           // stride hi16, dim1_stride lo
  g1[7] = 0;
  v4i z4 = {0, 0, 0, 0};
#if TDM_6ARG
  v8i z8 = {0, 0, 0, 0, 0, 0, 0, 0};
  __builtin_amdgcn_tensor_load_to_lds(g0, g1, z4, z4, z8, 0);
#else
  __builtin_amdgcn_tensor_load_to_lds(g0, g1, z4, z4, 0);
#endif
}
#endif

// ---------------- conversion kernels ----------------
__global__ void cvt_bf16_kernel(const float* __restrict__ src,
                                unsigned short* __restrict__ dst, long n) {
  long i = (long)blockIdx.x * blockDim.x + threadIdx.x;
  long stride = (long)gridDim.x * blockDim.x;
  for (; i < n; i += stride) dst[i] = f2bf(src[i]);
}

// Transpose-convert one (E x E) f32 weight [k][n] into bf16 dst[n+rowOff][k].
__global__ __launch_bounds__(256)
void cvtT_kernel(const float* __restrict__ w, unsigned short* __restrict__ dst,
                 int rowOff, int ldd) {
  __shared__ unsigned short tile[64][65];
  const int bn = blockIdx.x * 64;       // n base
  const int bk = blockIdx.y * 64;       // k base
  const int tid = threadIdx.x;
  const int tr = tid >> 2;              // 0..63
  const int tc = (tid & 3) * 16;        // {0,16,32,48}
#pragma unroll
  for (int j = 0; j < 16; ++j)
    tile[tr][tc + j] = f2bf(w[(size_t)(bk + tr) * E_DIM + bn + tc + j]);
  __syncthreads();
  unsigned short* dp = dst + (size_t)(rowOff + bn + tr) * ldd + bk + tc;
#pragma unroll
  for (int j = 0; j < 16; ++j) dp[j] = tile[tc + j][tr];
}

// ---------------- WMMA GEMM: C(f32, MxN) = A(bf16, MxK) * B^T(bf16, NxK) --
#define BM 128
#define BN 128
#define BK 32

__global__ __launch_bounds__(256)
void gemm_bf16f32_tn(const unsigned short* __restrict__ A,   // M x K row-major
                     const unsigned short* __restrict__ Bt,  // N x K row-major
                     float* __restrict__ C, int M, int N, int K) {
  __shared__ unsigned short As[2][BM * BK];
  __shared__ unsigned short Bs[2][BN * BK];
  (void)M;

  const int tid  = threadIdx.x;
  const int lane = tid & 31;
  const int wave = tid >> 5;        // 0..7
  const int wm   = wave >> 2;       // 0..1  -> 64-row slab
  const int wn   = wave & 3;        // 0..3  -> 32-col slab
  const int bm   = blockIdx.y * BM;
  const int bn   = blockIdx.x * BN;
  const int half = lane >> 4;       // K-half selector
  const int l16  = lane & 15;
  const int nk   = K / BK;

  v8f acc[4][2];
#pragma unroll
  for (int i = 0; i < 4; ++i)
#pragma unroll
    for (int j = 0; j < 2; ++j) {
      v8f z = {0.f,0.f,0.f,0.f,0.f,0.f,0.f,0.f};
      acc[i][j] = z;
    }

#if TDM_AVAIL
  // ---- TDM-staged, double-buffered pipeline ----
  const unsigned asOff[2] = {(unsigned)(unsigned long long)(const void*)&As[0][0],
                             (unsigned)(unsigned long long)(const void*)&As[1][0]};
  const unsigned bsOff[2] = {(unsigned)(unsigned long long)(const void*)&Bs[0][0],
                             (unsigned)(unsigned long long)(const void*)&Bs[1][0]};
  const unsigned strideB = (unsigned)K * 2u;
  const unsigned short* Abase = A  + (size_t)bm * K;
  const unsigned short* Bbase = Bt + (size_t)bn * K;

  if (wave == 0) {
    tdm_load_tile(asOff[0], Abase, BK * 2, BM, strideB);
    tdm_load_tile(bsOff[0], Bbase, BK * 2, BN, strideB);
  }
  for (int i = 0; i < nk; ++i) {
    const int cur = i & 1;
    if (wave == 0) {
      if (i + 1 < nk) {
        tdm_load_tile(asOff[cur ^ 1], Abase + (size_t)(i + 1) * BK, BK * 2, BM, strideB);
        tdm_load_tile(bsOff[cur ^ 1], Bbase + (size_t)(i + 1) * BK, BK * 2, BN, strideB);
        __builtin_amdgcn_s_wait_tensorcnt(2);   // tile i fully landed
      } else {
        __builtin_amdgcn_s_wait_tensorcnt(0);
      }
    }
    __syncthreads();
    {
      const unsigned short* as = As[cur];
      const unsigned short* bs = Bs[cur];
      v16bf af[4], bfg[2];
#pragma unroll
      for (int ii = 0; ii < 4; ++ii) {
        const unsigned short* p = &as[(wm * 64 + ii * 16 + l16) * BK + half * 8];
        *((uint4*)(&af[ii]))     = *(const uint4*)(p);
        *((uint4*)(&af[ii]) + 1) = *(const uint4*)(p + 16);
      }
#pragma unroll
      for (int jj = 0; jj < 2; ++jj) {
        const unsigned short* p = &bs[(wn * 32 + jj * 16 + l16) * BK + half * 8];
        *((uint4*)(&bfg[jj]))     = *(const uint4*)(p);
        *((uint4*)(&bfg[jj]) + 1) = *(const uint4*)(p + 16);
      }
#pragma unroll
      for (int ii = 0; ii < 4; ++ii)
#pragma unroll
        for (int jj = 0; jj < 2; ++jj)
          acc[ii][jj] = __builtin_amdgcn_wmma_f32_16x16x32_bf16(
              false, af[ii], false, bfg[jj], (short)0, acc[ii][jj], false, false);
    }
    __syncthreads();
  }
#else
  // ---- fallback: register-staged b128 loads + LDS stores (single buffer) --
  const int ra = tid >> 1;                 // row within tile 0..127
  const int ka = (tid & 1) * 16;           // k-chunk {0,16}
  const unsigned short* Ag = A  + (size_t)(bm + ra) * K + ka;
  const unsigned short* Bg = Bt + (size_t)(bn + ra) * K + ka;
  for (int k0 = 0; k0 < K; k0 += BK) {
    uint4 a0 = *(const uint4*)(Ag);
    uint4 a1 = *(const uint4*)(Ag + 8);
    uint4 b0 = *(const uint4*)(Bg);
    uint4 b1 = *(const uint4*)(Bg + 8);
    *(uint4*)(&As[0][ra * BK + ka])     = a0;
    *(uint4*)(&As[0][ra * BK + ka + 8]) = a1;
    *(uint4*)(&Bs[0][ra * BK + ka])     = b0;
    *(uint4*)(&Bs[0][ra * BK + ka + 8]) = b1;
    __builtin_prefetch(Ag + BK, 0, 1);
    __builtin_prefetch(Bg + BK, 0, 1);
    __syncthreads();
    {
      v16bf af[4], bfg[2];
#pragma unroll
      for (int ii = 0; ii < 4; ++ii) {
        const unsigned short* p = &As[0][(wm * 64 + ii * 16 + l16) * BK + half * 8];
        *((uint4*)(&af[ii]))     = *(const uint4*)(p);
        *((uint4*)(&af[ii]) + 1) = *(const uint4*)(p + 16);
      }
#pragma unroll
      for (int jj = 0; jj < 2; ++jj) {
        const unsigned short* p = &Bs[0][(wn * 32 + jj * 16 + l16) * BK + half * 8];
        *((uint4*)(&bfg[jj]))     = *(const uint4*)(p);
        *((uint4*)(&bfg[jj]) + 1) = *(const uint4*)(p + 16);
      }
#pragma unroll
      for (int ii = 0; ii < 4; ++ii)
#pragma unroll
        for (int jj = 0; jj < 2; ++jj)
          acc[ii][jj] = __builtin_amdgcn_wmma_f32_16x16x32_bf16(
              false, af[ii], false, bfg[jj], (short)0, acc[ii][jj], false, false);
    }
    __syncthreads();
    Ag += BK;
    Bg += BK;
  }
#endif

  // write back: v8f element r -> row 8*half + r, col l16 of each 16x16 tile
#pragma unroll
  for (int i = 0; i < 4; ++i)
#pragma unroll
    for (int j = 0; j < 2; ++j) {
      int row0 = bm + wm * 64 + i * 16 + half * 8;
      int col  = bn + wn * 32 + j * 16 + l16;
      float* Cp = C + (size_t)row0 * N + col;
#pragma unroll
      for (int r = 0; r < 8; ++r) Cp[(size_t)r * N] = acc[i][j][r];
    }
}

// ---------------- block 1 attention (per (n,l): 12x12 head mixing) --------
__global__ __launch_bounds__(256)
void attn1_kernel(const float* __restrict__ qkv, unsigned short* __restrict__ attnb) {
  __shared__ float q[E_DIM], k[E_DIM], v[E_DIM];
  __shared__ float en[H_NUM * H_NUM];
  __shared__ float pr[H_NUM * H_NUM];
  const int s = blockIdx.x;                 // n*L + l
  const int tid = threadIdx.x;
  const float* base = qkv + (size_t)s * N3;
  for (int i = tid; i < E_DIM; i += 256) {
    q[i] = base[i];
    k[i] = base[E_DIM + i];
    v[i] = base[2 * E_DIM + i];
  }
  __syncthreads();
  const float inv_sqrt_e = 0.0255155180f;   // 1/sqrt(1536)
  if (tid < H_NUM * H_NUM) {
    int h1 = tid / H_NUM, h2 = tid % H_NUM;
    const float* qp = q + h1 * D_DIM;
    const float* kp = k + h2 * D_DIM;
    float d0 = 0.f;
    for (int j = 0; j < D_DIM; ++j) d0 += qp[j] * kp[j];
    en[tid] = d0 * inv_sqrt_e;
  }
  __syncthreads();
  if (tid < H_NUM) {
    float mx = -1e30f;
    for (int j = 0; j < H_NUM; ++j) mx = fmaxf(mx, en[tid * H_NUM + j]);
    float tmp[H_NUM], sum = 0.f;
    for (int j = 0; j < H_NUM; ++j) { tmp[j] = __expf(en[tid * H_NUM + j] - mx); sum += tmp[j]; }
    float r = 1.0f / sum;
    for (int j = 0; j < H_NUM; ++j) pr[tid * H_NUM + j] = tmp[j] * r;
  }
  __syncthreads();
  unsigned short* outp = attnb + (size_t)s * E_DIM;
  for (int e = tid; e < E_DIM; e += 256) {
    int h = e >> 7, d = e & 127;
    float a = 0.f;
    for (int h2 = 0; h2 < H_NUM; ++h2) a += pr[h * H_NUM + h2] * v[h2 * D_DIM + d];
    outp[e] = f2bf(a);
  }
}

// ---------------- bias + residual + layernorm per row ---------------------
__global__ __launch_bounds__(256)
void ln_kernel(const float* __restrict__ obuf, const float* __restrict__ x,
               const float* __restrict__ bo, const float* __restrict__ gamma,
               const float* __restrict__ beta, float* __restrict__ out2) {
  __shared__ float y[E_DIM];
  __shared__ float red[256], red2[256];
  const int s = blockIdx.x, tid = threadIdx.x;
  const float* op = obuf + (size_t)s * E_DIM;
  const float* xp = x + (size_t)s * E_DIM;
  float sum = 0.f, sq = 0.f;
  for (int i = tid; i < E_DIM; i += 256) {
    float t = op[i] + bo[i] + xp[i];
    y[i] = t; sum += t; sq += t * t;
  }
  red[tid] = sum; red2[tid] = sq;
  __syncthreads();
  for (int st = 128; st > 0; st >>= 1) {
    if (tid < st) { red[tid] += red[tid + st]; red2[tid] += red2[tid + st]; }
    __syncthreads();
  }
  float mu  = red[0] * (1.0f / E_DIM);
  float var = red2[0] * (1.0f / E_DIM) - mu * mu;
  float rs  = rsqrtf(var + 1e-5f);
  float* o2 = out2 + (size_t)s * E_DIM;
  for (int i = tid; i < E_DIM; i += 256)
    o2[i] = gamma[i] * (y[i] - mu) * rs + beta[i];
}

// ---------------- e_state = mean over l of out2 ---------------------------
__global__ void estate_kernel(const float* __restrict__ out2, float* __restrict__ est) {
  int idx = blockIdx.x * 256 + threadIdx.x;         // 0..16*1536-1
  if (idx >= N_B * E_DIM) return;
  int n = idx / E_DIM, e = idx % E_DIM;
  const float* p = out2 + (size_t)n * L_SEQ * E_DIM + e;
  float s = 0.f;
  for (int l = 0; l < L_SEQ; ++l) s += p[(size_t)l * E_DIM];
  est[idx] = s * (1.0f / L_SEQ);
}

// ---------------- second block: raw scores --------------------------------
__global__ __launch_bounds__(256)
void scores_kernel(const float* __restrict__ out2, const float* __restrict__ est,
                   float* __restrict__ wraw) {
  __shared__ float row[E_DIM];
  __shared__ float part[256];
  const int t = blockIdx.x;        // 0..1021
  const int b = blockIdx.y;        // 0..15
  const int tid = threadIdx.x;
  const float* rp = out2 + ((size_t)b * L_SEQ + t + 2) * E_DIM;
  for (int i = tid; i < E_DIM; i += 256) row[i] = rp[i];
  __syncthreads();
  const int i = tid >> 4;          // state row
  const int j = tid & 15;
  const float* sp = est + i * E_DIM;
  float a = 0.f;
  for (int e = j; e < E_DIM; e += 16) a += sp[e] * row[e];
  part[tid] = a;
  __syncthreads();
  if (j < 8) part[tid] += part[tid + 8];
  __syncthreads();
  if (j < 4) part[tid] += part[tid + 4];
  __syncthreads();
  if (j < 2) part[tid] += part[tid + 2];
  __syncthreads();
  if (j == 0)
    wraw[((size_t)b * 16 + i) * 1022 + t] = (part[tid] + part[tid + 1]) * 0.0255155180f;
}

// ---------------- second block: softmax over t, in place ------------------
__global__ __launch_bounds__(256)
void softmaxw_kernel(float* __restrict__ w) {
  __shared__ float red[256];
  const int r = blockIdx.x;        // b*16+i
  float* p = w + (size_t)r * 1022;
  const int tid = threadIdx.x;
  float mx = -1e30f;
  for (int t = tid; t < 1022; t += 256) mx = fmaxf(mx, p[t]);
  red[tid] = mx; __syncthreads();
  for (int st = 128; st > 0; st >>= 1) {
    if (tid < st) red[tid] = fmaxf(red[tid], red[tid + st]);
    __syncthreads();
  }
  mx = red[0];
  __syncthreads();
  float sum = 0.f;
  for (int t = tid; t < 1022; t += 256) { float e = __expf(p[t] - mx); p[t] = e; sum += e; }
  red[tid] = sum; __syncthreads();
  for (int st = 128; st > 0; st >>= 1) {
    if (tid < st) red[tid] += red[tid + st];
    __syncthreads();
  }
  float inv = 1.0f / red[0];
  for (int t = tid; t < 1022; t += 256) p[t] *= inv;
}

// ---------------- second block: attended = weights @ e_actions ------------
__global__ __launch_bounds__(256)
void attended_kernel(const float* __restrict__ out2, const float* __restrict__ w,
                     float* __restrict__ att) {
  __shared__ float ws[1022];
  const int r = blockIdx.x;        // b*16+i
  const int b = r >> 4;
  const int tid = threadIdx.x;
  const float* wp = w + (size_t)r * 1022;
  for (int t = tid; t < 1022; t += 256) ws[t] = wp[t];
  __syncthreads();
  float acc[6] = {0.f, 0.f, 0.f, 0.f, 0.f, 0.f};
  const float* base = out2 + (size_t)b * L_SEQ * E_DIM + 2 * E_DIM;
  for (int t = 0; t < 1022; ++t) {
    float wv = ws[t];
    const float* rp = base + (size_t)t * E_DIM;
#pragma unroll
    for (int j = 0; j < 6; ++j) acc[j] += wv * rp[tid + j * 256];
  }
  float* op = att + (size_t)r * E_DIM;
#pragma unroll
  for (int j = 0; j < 6; ++j) op[tid + j * 256] = acc[j];
}

// ---------------------------------------------------------------------------
extern "C" void kernel_launch(void* const* d_in, const int* in_sizes, int n_in,
                              void* d_out, int out_size, void* d_ws, size_t ws_size,
                              hipStream_t stream) {
  (void)in_sizes; (void)n_in; (void)out_size; (void)ws_size;
  const float* x     = (const float*)d_in[0];
  const float* Wq    = (const float*)d_in[1];
  const float* Wk    = (const float*)d_in[2];
  const float* Wv    = (const float*)d_in[3];
  const float* Wo    = (const float*)d_in[4];
  const float* bo    = (const float*)d_in[5];
  const float* gamma = (const float*)d_in[6];
  const float* beta  = (const float*)d_in[7];

  char* ws = (char*)d_ws;
  unsigned short* xb    = (unsigned short*)(ws + OFF_XB);
  unsigned short* wcatT = (unsigned short*)(ws + OFF_WCAT);  // 4608 x 1536
  unsigned short* woT   = (unsigned short*)(ws + OFF_WOB);   // 1536 x 1536
  float*          qkv   = (float*)(ws + OFF_QKV);
  unsigned short* attnb = (unsigned short*)(ws + OFF_ATTNB);
  float*          obuf  = (float*)(ws + OFF_OBUF);
  float*          out2  = (float*)(ws + OFF_OUT2);
  float*          est   = (float*)(ws + OFF_EST);

  float* attended = (float*)d_out;                          // (1,16,16,1536)
  float* weights  = (float*)d_out + N_B * N_B * E_DIM;      // (1,16,16,1022)

  // 1. bf16 conversions: x plain; weights transposed (so GEMM B is N x K)
  cvt_bf16_kernel<<<4096, 256, 0, stream>>>(x, xb, (long)M_ROWS * E_DIM);
  dim3 gT(E_DIM / 64, E_DIM / 64);
  cvtT_kernel<<<gT, 256, 0, stream>>>(Wq, wcatT, 0,         E_DIM);
  cvtT_kernel<<<gT, 256, 0, stream>>>(Wk, wcatT, E_DIM,     E_DIM);
  cvtT_kernel<<<gT, 256, 0, stream>>>(Wv, wcatT, 2 * E_DIM, E_DIM);
  cvtT_kernel<<<gT, 256, 0, stream>>>(Wo, woT,   0,         E_DIM);

  // 2. fused QKV GEMM: (16384x1536) * (1536x4608) -> fp32
  dim3 gQKV(N3 / BN, M_ROWS / BM);
  gemm_bf16f32_tn<<<gQKV, 256, 0, stream>>>(xb, wcatT, qkv, M_ROWS, N3, E_DIM);

  // 3. head-mixing attention per (n,l); writes bf16 A-operand for Wo GEMM
  attn1_kernel<<<M_ROWS, 256, 0, stream>>>(qkv, attnb);

  // 4. output projection GEMM
  dim3 gO(E_DIM / BN, M_ROWS / BM);
  gemm_bf16f32_tn<<<gO, 256, 0, stream>>>(attnb, woT, obuf, M_ROWS, E_DIM, E_DIM);

  // 5. bias + residual + layernorm
  ln_kernel<<<M_ROWS, 256, 0, stream>>>(obuf, x, bo, gamma, beta, out2);

  // 6. e_state mean
  estate_kernel<<<(N_B * E_DIM + 255) / 256, 256, 0, stream>>>(out2, est);

  // 7. second SDPA block
  dim3 gS(1022, N_B);
  scores_kernel<<<gS, 256, 0, stream>>>(out2, est, weights);
  softmaxw_kernel<<<N_B * N_B, 256, 0, stream>>>(weights);
  attended_kernel<<<N_B * N_B, 256, 0, stream>>>(out2, weights, attended);
}
